// MultiHeadSelfAttention_33380485824770
// MI455X (gfx1250) — compile-verified, matrix operand order checked by reading, not run on hardware
//
#include <hip/hip_runtime.h>

// ---------------------------------------------------------------------------
// MHSA for MI455X (gfx1250, wave32, WMMA + async-to-LDS).
// B=2, S=2048, D=1024, H=16, P=64.
// ---------------------------------------------------------------------------

typedef __attribute__((ext_vector_type(16))) _Float16 v16h;
typedef __attribute__((ext_vector_type(8)))  float    v8f;

#define S_LEN   2048
#define D_DIM   1024
#define H_NUM   16
#define P_DIM   64
#define M_ROWS  4096   // B * S

union FragH { v16h v; uint4 q[2]; _Float16 h[16]; };

// Async global->LDS copy, 16 bytes per lane (GLOBAL_LOAD_ASYNC_TO_LDS_B128,
// tracked by ASYNCcnt).  LDS offset = low 32 bits of the generic pointer
__device__ __forceinline__ void async_copy_b128(void* lds_ptr, const void* gptr) {
  asm volatile("global_load_async_to_lds_b128 %0, %1, off"
               :
               : "v"((unsigned int)(uintptr_t)lds_ptr),
                 "v"((unsigned long long)(uintptr_t)gptr)
               : "memory");
}
__device__ __forceinline__ void wait_async0() {
#if __has_builtin(__builtin_amdgcn_s_wait_asynccnt)
  __builtin_amdgcn_s_wait_asynccnt(0);
#else
  asm volatile("s_wait_asynccnt 0" ::: "memory");
#endif
}

// Pack two f32 -> two f16 with v_cvt_pk_rtz_f16_f32 and store as one b32.
__device__ __forceinline__ void store_pk(_Float16* dst, float x, float y) {
  auto p = __builtin_amdgcn_cvt_pkrtz(x, y);
  *(decltype(p)*)dst = p;
}

// Load a 16x32 f16 fragment from an LDS tile (row-major for A, N-major for B)
// with the given stride.  Matches the CDNA5 16-bit A layout: lanes 0-15 hold
// K {0..7,16..23}, lanes 16-31 hold K {8..15,24..31}.
__device__ __forceinline__ v16h ldfrag(const _Float16* lds, int row0, int k0,
                                       int stride, int lane) {
  FragH f;
  const _Float16* p = lds + (size_t)(row0 + (lane & 15)) * stride + k0 + ((lane >> 4) << 3);
  f.q[0] = *(const uint4*)(p);
  f.q[1] = *(const uint4*)(p + 16);
  return f.v;
}

__device__ __forceinline__ v8f wmma_f16(v16h a, v16h b, v8f c) {
  return __builtin_amdgcn_wmma_f32_16x16x32_f16(false, a, false, b, (short)0, c,
                                                false, false);
}

// ---------------------------------------------------------------------------
// Generic tiled GEMM:  C[M,N] = A[M,K] * W[K,N] + bias[N]
//   IN_F16  : A is fp16 workspace (staged via async-to-LDS)
//   OUT_QKV : write fp16 into [B,H,S,P] workspace layout; else fp32 row-major
// Block = 256 threads (8 waves, 4x2), tile 128x128, K-slab 64 (16 WMMAs per
// barrier pair).
// ---------------------------------------------------------------------------
template <bool IN_F16, bool OUT_QKV>
__global__ __launch_bounds__(256) void gemm_wmma_kernel(
    const void* __restrict__ Aptr, const float* __restrict__ W,
    const float* __restrict__ bias, void* __restrict__ Cptr,
    int M, int N, int Kdim) {
  __shared__ __attribute__((aligned(16))) _Float16 As[128 * 64];  // row-major
  __shared__ __attribute__((aligned(16))) _Float16 Bs[128 * 64];  // N-major

  const int tid  = threadIdx.x;
  const int wave = tid >> 5, lane = tid & 31;
  const int wm = wave & 3, wn = wave >> 2;
  const int lcol = lane & 15, kg = lane >> 4;

  const int bn = blockIdx.x * 128;
  const int bm = blockIdx.y * 128;

  const int arow  = tid >> 1;          // 0..127
  const int ahalf = (tid & 1) * 32;    // K offset 0 or 32
  const int bkrow = tid >> 2;          // 0..63  (K row of W)
  const int bng   = (tid & 3) * 32;    // N offset

  v8f acc[2][4];
#pragma unroll
  for (int mt = 0; mt < 2; ++mt)
#pragma unroll
    for (int nt = 0; nt < 4; ++nt)
#pragma unroll
      for (int r = 0; r < 8; ++r) acc[mt][nt][r] = 0.0f;

  for (int k0 = 0; k0 < Kdim; k0 += 64) {
    // ---- stage A (128 x 64) ----
    if constexpr (IN_F16) {
      const _Float16* src = (const _Float16*)Aptr + (size_t)(bm + arow) * Kdim + k0 + ahalf;
      _Float16* dst = As + arow * 64 + ahalf;
#pragma unroll
      for (int i = 0; i < 4; ++i)
        async_copy_b128(dst + i * 8, src + i * 8);
    } else {
      const float* src = (const float*)Aptr + (size_t)(bm + arow) * Kdim + k0 + ahalf;
      _Float16* dst = As + arow * 64 + ahalf;
#pragma unroll
      for (int i = 0; i < 32; i += 4) {
        float4 f = *(const float4*)(src + i);
        store_pk(dst + i, f.x, f.y);
        store_pk(dst + i + 2, f.z, f.w);
      }
    }
    // ---- stage W (64 x 128) transposed into N-major LDS ----
    {
      const float* wsrc = W + (size_t)(k0 + bkrow) * N + bn + bng;
#pragma unroll
      for (int j = 0; j < 32; ++j)
        Bs[(bng + j) * 64 + bkrow] = (_Float16)wsrc[j];
    }
    if constexpr (IN_F16) wait_async0();
    __syncthreads();

    if (k0 + 64 < Kdim)  // hint next weight tile toward L2
      __builtin_prefetch(W + (size_t)(k0 + 64 + bkrow) * N + bn + bng, 0, 1);

#pragma unroll
    for (int pk = 0; pk < 64; pk += 32) {
      v16h af[2], bf[4];
#pragma unroll
      for (int mt = 0; mt < 2; ++mt) af[mt] = ldfrag(As, wm * 32 + mt * 16, pk, 64, lane);
#pragma unroll
      for (int nt = 0; nt < 4; ++nt) bf[nt] = ldfrag(Bs, wn * 64 + nt * 16, pk, 64, lane);
#pragma unroll
      for (int mt = 0; mt < 2; ++mt)
#pragma unroll
        for (int nt = 0; nt < 4; ++nt)
          acc[mt][nt] = wmma_f16(af[mt], bf[nt], acc[mt][nt]);
    }
    __syncthreads();
  }

  // ---- epilogue: bias + store.  D layout: row = r + 8*kg, col = lcol ----
#pragma unroll
  for (int mt = 0; mt < 2; ++mt)
#pragma unroll
    for (int nt = 0; nt < 4; ++nt) {
      const int n  = bn + wn * 64 + nt * 16 + lcol;
      const float bv = bias[n];
#pragma unroll
      for (int r = 0; r < 8; ++r) {
        const int m = bm + wm * 32 + mt * 16 + r + kg * 8;
        const float val = acc[mt][nt][r] + bv;
        if constexpr (OUT_QKV) {
          const int b = m >> 11;        // m / S
          const int s = m & (S_LEN - 1);
          const int h = n >> 6;         // n / P
          const int p = n & (P_DIM - 1);
          ((_Float16*)Cptr)[((((size_t)b * H_NUM + h) * S_LEN + s) << 6) + p] =
              (_Float16)val;
        } else {
          ((float*)Cptr)[(size_t)m * N + n] = val;
        }
      }
    }
}

// ---------------------------------------------------------------------------
// Causal flash attention.  Q,K,V fp16 in [B,H,S,P]; output fp16 in [B,S,H*P].
// Block = 256 threads handles one (b,h) x 128 query rows; key tiles of 64.
// Q/K tiles staged with GLOBAL_LOAD_ASYNC_TO_LDS_B128; V transposed via VGPRs.
// ---------------------------------------------------------------------------
__global__ __launch_bounds__(256) void flash_attn_kernel(
    const _Float16* __restrict__ Qg, const _Float16* __restrict__ Kg,
    const _Float16* __restrict__ Vg, _Float16* __restrict__ Og, float scale) {
  __shared__ __attribute__((aligned(16))) _Float16 sQ[128 * 64];   // [q][p]
  __shared__ __attribute__((aligned(16))) _Float16 sK[64 * 64];    // [t][p]
  __shared__ __attribute__((aligned(16))) _Float16 sVT[64 * 64];   // [p][t]
  __shared__ __attribute__((aligned(16))) _Float16 sP[128 * 64];   // [q][t]
  __shared__ float rowM[128], rowL[128], pM[2][128], pL[2][128];

  const int tid  = threadIdx.x;
  const int wave = tid >> 5, lane = tid & 31;
  const int wm = wave & 3, wn = wave >> 2;
  const int lcol = lane & 15, kg = lane >> 4;

  const int qb = blockIdx.x;        // query block (0..15)
  const int bh = blockIdx.y;        // b*H + h   (0..31)
  const size_t head = (size_t)bh * S_LEN * P_DIM;

  // ---- stage Q tile (128x64) asynchronously ----
  {
    const int row = tid >> 1, half = (tid & 1) * 32;
    const _Float16* src = Qg + head + (size_t)(qb * 128 + row) * P_DIM + half;
    _Float16* dst = sQ + row * 64 + half;
#pragma unroll
    for (int i = 0; i < 4; ++i)
      async_copy_b128(dst + i * 8, src + i * 8);
  }
  if (tid < 128) { rowM[tid] = -1e30f; rowL[tid] = 0.0f; }

  v8f oacc[2][2];
#pragma unroll
  for (int mt = 0; mt < 2; ++mt)
#pragma unroll
    for (int np = 0; np < 2; ++np)
#pragma unroll
      for (int r = 0; r < 8; ++r) oacc[mt][np][r] = 0.0f;

  const int ktmax = 2 * qb + 1;     // causal: keys up to qb*128+127
  for (int kt = 0; kt <= ktmax; ++kt) {
    const int t0 = kt * 64;
    // ---- stage K tile (64x64) async; V tile transposed via VGPRs ----
    {
      const int row = tid >> 2, q4 = (tid & 3) * 16;
      const _Float16* ks = Kg + head + (size_t)(t0 + row) * P_DIM + q4;
      _Float16* kd = sK + row * 64 + q4;
      async_copy_b128(kd, ks);
      async_copy_b128(kd + 8, ks + 8);
      FragH vb;
      const uint4* vs = (const uint4*)(Vg + head + (size_t)(t0 + row) * P_DIM + q4);
      vb.q[0] = vs[0]; vb.q[1] = vs[1];
#pragma unroll
      for (int j = 0; j < 16; ++j) sVT[(q4 + j) * 64 + row] = vb.h[j];
    }
    wait_async0();     // covers Q (first iter) + this K tile
    __syncthreads();

    // ---- scores: S = Q * K^T (per wave 32x32, K-dim = P = 64) ----
    v8f sc[2][2];
#pragma unroll
    for (int mt = 0; mt < 2; ++mt)
#pragma unroll
      for (int nt = 0; nt < 2; ++nt)
#pragma unroll
        for (int r = 0; r < 8; ++r) sc[mt][nt][r] = 0.0f;
#pragma unroll
    for (int pk = 0; pk < 64; pk += 32) {
      v16h aq[2], bk[2];
#pragma unroll
      for (int mt = 0; mt < 2; ++mt) aq[mt] = ldfrag(sQ, wm * 32 + mt * 16, pk, 64, lane);
#pragma unroll
      for (int nt = 0; nt < 2; ++nt) bk[nt] = ldfrag(sK, wn * 32 + nt * 16, pk, 64, lane);
#pragma unroll
      for (int mt = 0; mt < 2; ++mt)
#pragma unroll
        for (int nt = 0; nt < 2; ++nt)
          sc[mt][nt] = wmma_f16(aq[mt], bk[nt], sc[mt][nt]);
    }

    // ---- scale + causal mask ----
#pragma unroll
    for (int mt = 0; mt < 2; ++mt)
#pragma unroll
      for (int nt = 0; nt < 2; ++nt)
#pragma unroll
        for (int r = 0; r < 8; ++r) {
          const int rg = qb * 128 + wm * 32 + mt * 16 + r + kg * 8;
          const int cg = t0 + wn * 32 + nt * 16 + lcol;
          const float v = sc[mt][nt][r] * scale;
          sc[mt][nt][r] = (cg > rg) ? -1e30f : v;
        }

    // ---- per-wave partial row max (reduce across 16 lanes sharing a row) ----
#pragma unroll
    for (int mt = 0; mt < 2; ++mt)
#pragma unroll
      for (int r = 0; r < 8; ++r) {
        float m = fmaxf(sc[mt][0][r], sc[mt][1][r]);
        m = fmaxf(m, __shfl_xor(m, 1, 32));
        m = fmaxf(m, __shfl_xor(m, 2, 32));
        m = fmaxf(m, __shfl_xor(m, 4, 32));
        m = fmaxf(m, __shfl_xor(m, 8, 32));
        if (lcol == 0) pM[wn][wm * 32 + mt * 16 + r + kg * 8] = m;
      }
    __syncthreads();

    // ---- p = exp(s - newM), partial row sums, rescale O by exp(oldM-newM) ----
#pragma unroll
    for (int mt = 0; mt < 2; ++mt)
#pragma unroll
      for (int r = 0; r < 8; ++r) {
        const int rl = wm * 32 + mt * 16 + r + kg * 8;
        const float oldM = rowM[rl];
        const float newM = fmaxf(oldM, fmaxf(pM[0][rl], pM[1][rl]));
        const float corr = __expf(oldM - newM);
        float psum = 0.0f;
#pragma unroll
        for (int nt = 0; nt < 2; ++nt) {
          const float pv = __expf(sc[mt][nt][r] - newM);
          psum += pv;
          sP[rl * 64 + wn * 32 + nt * 16 + lcol] = (_Float16)pv;
        }
        psum += __shfl_xor(psum, 1, 32);
        psum += __shfl_xor(psum, 2, 32);
        psum += __shfl_xor(psum, 4, 32);
        psum += __shfl_xor(psum, 8, 32);
        if (lcol == 0) pL[wn][rl] = psum;
        oacc[mt][0][r] *= corr;
        oacc[mt][1][r] *= corr;
      }
    __syncthreads();

    // rowM/rowL update overlaps the PV WMMAs; next read of rowM/rowL is
    // fenced by the stage barrier of the next iteration (or loop exit).
    if (tid < 128) {
      const float oldM = rowM[tid];
      const float newM = fmaxf(oldM, fmaxf(pM[0][tid], pM[1][tid]));
      rowL[tid] = rowL[tid] * __expf(oldM - newM) + pL[0][tid] + pL[1][tid];
      rowM[tid] = newM;
    }

    // ---- O += P * V  (per wave 32x32 of the 128x64 output, K-dim = 64) ----
#pragma unroll
    for (int tk = 0; tk < 64; tk += 32) {
      v16h ap[2], bv[2];
#pragma unroll
      for (int mt = 0; mt < 2; ++mt) ap[mt] = ldfrag(sP, wm * 32 + mt * 16, tk, 64, lane);
#pragma unroll
      for (int np = 0; np < 2; ++np) bv[np] = ldfrag(sVT, wn * 32 + np * 16, tk, 64, lane);
#pragma unroll
      for (int mt = 0; mt < 2; ++mt)
#pragma unroll
        for (int np = 0; np < 2; ++np)
          oacc[mt][np] = wmma_f16(ap[mt], bv[np], oacc[mt][np]);
    }
    __syncthreads();
  }

  // ---- epilogue: O *= 1/l, store fp16 in concat layout [B,S,H*P] ----
  const int b = bh >> 4, h = bh & 15;
#pragma unroll
  for (int mt = 0; mt < 2; ++mt)
#pragma unroll
    for (int np = 0; np < 2; ++np) {
      const int p = wn * 32 + np * 16 + lcol;
#pragma unroll
      for (int r = 0; r < 8; ++r) {
        const int rl = wm * 32 + mt * 16 + r + kg * 8;
        const int s  = qb * 128 + rl;
        const float val = oacc[mt][np][r] * __builtin_amdgcn_rcpf(rowL[rl]);
        Og[(size_t)(b * S_LEN + s) * D_DIM + h * P_DIM + p] = (_Float16)val;
      }
    }
}

// ---------------------------------------------------------------------------
extern "C" void kernel_launch(void* const* d_in, const int* in_sizes, int n_in,
                              void* d_out, int out_size, void* d_ws, size_t ws_size,
                              hipStream_t stream) {
  const float* X  = (const float*)d_in[0];
  const float* Wq = (const float*)d_in[1];
  const float* bq = (const float*)d_in[2];
  const float* Wk = (const float*)d_in[3];
  const float* bk = (const float*)d_in[4];
  const float* Wv = (const float*)d_in[5];
  const float* bv = (const float*)d_in[6];
  const float* Wo = (const float*)d_in[7];
  const float* bo = (const float*)d_in[8];

  const size_t plane = (size_t)M_ROWS * D_DIM;  // 4M fp16 elems per tensor
  _Float16* qws = (_Float16*)d_ws;
  _Float16* kws = qws + plane;
  _Float16* vws = kws + plane;
  _Float16* ows = vws + plane;

  dim3 gGemm(D_DIM / 128, M_ROWS / 128);  // (8, 32)
  dim3 blk(256);

  // Q/K/V projections: fp32 -> fp16 [B,H,S,P]
  gemm_wmma_kernel<false, true><<<gGemm, blk, 0, stream>>>(X, Wq, bq, qws,
                                                           M_ROWS, D_DIM, D_DIM);
  gemm_wmma_kernel<false, true><<<gGemm, blk, 0, stream>>>(X, Wk, bk, kws,
                                                           M_ROWS, D_DIM, D_DIM);
  gemm_wmma_kernel<false, true><<<gGemm, blk, 0, stream>>>(X, Wv, bv, vws,
                                                           M_ROWS, D_DIM, D_DIM);

  // Causal flash attention, scale = 1/sqrt(P) = 0.125
  dim3 gAttn(S_LEN / 128, 2 * H_NUM);  // (16, 32)
  flash_attn_kernel<<<gAttn, blk, 0, stream>>>(qws, kws, vws, ows, 0.125f);

  // Output projection: fp16 attn @ Wo + bo -> fp32 d_out
  gemm_wmma_kernel<true, false><<<gGemm, blk, 0, stream>>>(ows, Wo, bo, d_out,
                                                           M_ROWS, D_DIM, D_DIM);
}
